// SimpleSchNetModel_37220186587472
// MI455X (gfx1250) — matrix operand order repaired and checked by reading, not compiled, not run on hardware
//
#include <hip/hip_runtime.h>
#include <hip/hip_bf16.h>

// ---- problem constants (match reference) ----
#define B_      16
#define N_      512
#define D_      512
#define V_      30
#define L_      4
#define RBF_    128
#define K_      32
#define BN_     (B_ * N_)       // 8192
#define E_      (BN_ * K_)      // 262144
#define CUTOFF_ 6.0f
#define GAMMA_  10.0f

typedef __attribute__((ext_vector_type(16))) _Float16 v16h;
typedef __attribute__((ext_vector_type(8)))  _Float16 v8h;
typedef __attribute__((ext_vector_type(2)))  _Float16 v2h;
typedef __attribute__((ext_vector_type(8)))  float    v8f;

union AFrag { v16h v; v8h h[2]; };

__device__ __forceinline__ v8f wmma_f16(v16h a, v16h b, v8f c) {
    // D = A(16x32 f16) * B(32x16 f16) + C(16x16 f32)
    return __builtin_amdgcn_wmma_f32_16x16x32_f16(
        /*neg_a=*/false, a, /*neg_b=*/false, b,
        /*c_mod=*/(short)0, c, /*reuse_a=*/false, /*reuse_b=*/false);
}

// Fast silu: v * sigmoid(v) using v_exp_f32 + v_rcp_f32 (no IEEE div sequence).
__device__ __forceinline__ float silu_f(float v) {
    return v * __builtin_amdgcn_rcpf(1.0f + __expf(-v));
}

// B fragment: packed layout, 32 lanes x 16 halves contiguous per fragment.
__device__ __forceinline__ v16h load_b_frag(const _Float16* __restrict__ packed,
                                            int frag, int lane) {
    return *(const v16h*)(packed + ((long)frag * 32 + lane) * 16);
}

// A fragment from row-major f16 matrix (global). ISA 16-bit A layout:
// lanes 0-15: row=lane, K = k0+0..7 and k0+16..23
// lanes 16-31: row=lane-16, K = k0+8..15 and k0+24..31
__device__ __forceinline__ v16h load_a_frag_g(const _Float16* __restrict__ rows,
                                              long stride_h, int k0, int lane) {
    const int r  = lane & 15;
    const int hi = lane >> 4;
    const _Float16* p = rows + (long)r * stride_h + k0 + hi * 8;
    AFrag u;
    u.h[0] = *(const v8h*)(p);
    u.h[1] = *(const v8h*)(p + 16);
    return u.v;
}

// ============================================================
// Weight pre-pack: f32 [layers][Kdim][Ncols] -> f16 fragment-major
// fragment index fi = (l*KS + ks)*NT + nt ; within: lane(0..31) x half(0..15)
//   k = ks*32 + (lane>>4)*8 + (h&7) + (h>>3)*16 ;  n = nt*16 + (lane&15)
// ============================================================
__global__ void pack_w_k(const float* __restrict__ W, _Float16* __restrict__ out,
                         int Kdim, int Ncols, long total) {
    long t = (long)blockIdx.x * blockDim.x + threadIdx.x;
    if (t >= total) return;
    const int  h    = (int)(t & 15);
    const int  lane = (int)((t >> 4) & 31);
    const long fi   = t >> 9;
    const int  NT   = Ncols >> 4;
    const int  KS   = Kdim >> 5;
    const int  nt   = (int)(fi % NT);
    const long tmp  = fi / NT;
    const int  ks   = (int)(tmp % KS);
    const int  l    = (int)(tmp / KS);
    const int  k    = ks * 32 + (lane >> 4) * 8 + (h & 7) + (h >> 3) * 16;
    const int  n    = nt * 16 + (lane & 15);
    out[t] = (_Float16)W[((long)l * Kdim + k) * Ncols + n];
}

// ============================================================
// x[i,d] = embed[tok[i], d]   (writes residual stream in d_out)
// ============================================================
__global__ void embed_k(const int* __restrict__ tok,
                        const float* __restrict__ emb,
                        float* __restrict__ x) {
    long t = (long)blockIdx.x * 256 + threadIdx.x;   // BN*D threads
    int i = (int)(t >> 9);
    int d = (int)(t & (D_ - 1));
    x[t] = emb[(long)tok[i] * D_ + d];
}

__global__ void mask_k(const int* __restrict__ tok, float* __restrict__ m) {
    int i = blockIdx.x * 256 + threadIdx.x;
    if (i < BN_) m[i] = (tok[i] == 0) ? 1.0f : 0.0f;
}

// ============================================================
// Edge distances: dist[e] = ||c[row[e]] - c[col[e]]||
// ============================================================
__global__ void dist_k(const int* __restrict__ ei,
                       const float* __restrict__ c,
                       float* __restrict__ dist) {
    long e = (long)blockIdx.x * 256 + threadIdx.x;   // E threads
    int a = ei[e];
    int b = ei[(long)E_ + e];
    float dx = c[a * 3 + 0] - c[b * 3 + 0];
    float dy = c[a * 3 + 1] - c[b * 3 + 1];
    float dz = c[a * 3 + 2] - c[b * 3 + 2];
    dist[e] = sqrtf(dx * dx + dy * dy + dz * dz);
}

// rbf[e, r] = exp(-gamma * (dist[e] - center_r)^2)  -> f16
__global__ void rbf_k(const float* __restrict__ dist,
                      _Float16* __restrict__ rbf) {
    long t = (long)blockIdx.x * 256 + threadIdx.x;   // E*RBF threads
    long e = t >> 7;
    int  r = (int)(t & (RBF_ - 1));
    float ctr = (CUTOFF_ / (float)(RBF_ - 1)) * (float)r;
    float d   = dist[e] - ctr;
    rbf[t] = (_Float16)__expf(-GAMMA_ * d * d);
}

// ============================================================
// Fused edge pipeline, one workgroup (8 waves) per node i:
//   edges e0..e0+31 all have row == i (fixed-degree neighbor list)
//   h  = silu(rbf[e0:e0+32] @ W1 + b1)      (32x512, in LDS, f16)
//   ef = h @ W2 + b2                        (32x512, f32 in accumulators)
//   agg[i,:] = sum_e ef[e,:] * x[col[e],:]  (local reduction, LDS atomics only)
// Wave w owns mt = w&1 (16 edges) and nt group (w>>1)*8 .. +7 (128 cols):
// A fragments are loaded once and reused across all 8 column tiles.
// ============================================================
__global__ __launch_bounds__(256, 1)
void edge_layer_k(const _Float16* __restrict__ rbf,   // [E, RBF]
                  const int* __restrict__ ei,         // [2, E]
                  const float* __restrict__ x,        // [BN, D]
                  const _Float16* __restrict__ w1p,   // packed, KS=4  NT=32
                  const float* __restrict__ b1,       // [D]
                  const _Float16* __restrict__ w2p,   // packed, KS=16 NT=32
                  const float* __restrict__ b2,       // [D]
                  _Float16* __restrict__ aggh)        // [BN, D] f16
{
    constexpr int HS = D_ + 16;                       // LDS row stride (halves), 16B aligned
    __shared__ _Float16 hbuf[32 * HS];                // 33 KB
    __shared__ float    aggs[D_];
    __shared__ int      scol[32];

    const int  node = blockIdx.x;
    const int  tid  = threadIdx.x;
    const int  lane = tid & 31;
    const int  wave = tid >> 5;
    const long e0   = (long)node * K_;

    if (tid < 32) scol[tid] = ei[(long)E_ + e0 + tid];
    aggs[tid] = 0.0f;
    aggs[tid + 256] = 0.0f;
    __syncthreads();

    const int r  = lane & 15;
    const int hi = lane >> 4;
    const int mt = wave & 1;              // which 16-edge half this wave owns
    const int ng = (wave >> 1) * 8;       // first of 8 column tiles
    const int mbase = mt * 16 + 8 * hi;   // D-fragment row base for this lane

    // ---- Phase 1: GEMM1 (32x512, K=128) + bias + silu -> hbuf ----
    {
        const _Float16* arows = rbf + (e0 + mt * 16) * RBF_;
        v16h afr[4];
#pragma unroll
        for (int ks = 0; ks < 4; ++ks)
            afr[ks] = load_a_frag_g(arows, RBF_, ks * 32, lane);

#pragma unroll
        for (int j = 0; j < 8; ++j) {
            const int nt = ng + j;
            v8f acc = {};
#pragma unroll
            for (int ks = 0; ks < 4; ++ks) {
                v16h b = load_b_frag(w1p, ks * 32 + nt, lane);
                acc = wmma_f16(afr[ks], b, acc);
            }
            const int n     = nt * 16 + r;
            const float bia = b1[n];
#pragma unroll
            for (int q = 0; q < 8; ++q) {
                float v = silu_f(acc[q] + bia);
                hbuf[(mbase + q) * HS + n] = (_Float16)v;
            }
        }
    }
    __syncthreads();

    // ---- Phase 2: GEMM2 (32x512, K=512) + bias, * x[col], reduce ----
    {
        v8f acc[8] = {};
        for (int ks = 0; ks < 16; ++ks) {
            const _Float16* p = &hbuf[(mt * 16 + r) * HS + ks * 32 + hi * 8];
            AFrag ua;
            ua.h[0] = *(const v8h*)(p);        // ds_load_b128
            ua.h[1] = *(const v8h*)(p + 16);   // ds_load_b128
#pragma unroll
            for (int j = 0; j < 8; ++j) {
                v16h b = load_b_frag(w2p, ks * 32 + ng + j, lane);
                acc[j] = wmma_f16(ua.v, b, acc[j]);
            }
        }
#pragma unroll
        for (int j = 0; j < 8; ++j) {
            const int n     = (ng + j) * 16 + r;
            const float bia = b2[n];
            float s = 0.0f;
#pragma unroll
            for (int q = 0; q < 8; ++q) {
                const int  cidx = scol[mbase + q];
                const float xv  = x[(long)cidx * D_ + n];
                s += (acc[j][q] + bia) * xv;
            }
            s += __shfl_xor(s, 16, 32);            // combine m 0..7 with 8..15
            if (lane < 16) atomicAdd(&aggs[n], s); // ds_add_f32 (mt halves)
        }
    }
    __syncthreads();

    // packed f16 writeback: 2 adjacent halves per thread -> 4B store
    {
        const int d2 = tid * 2;
        v2h o;
        o.x = (_Float16)aggs[d2];
        o.y = (_Float16)aggs[d2 + 1];
        *(v2h*)(aggh + (long)node * D_ + d2) = o;
    }
}

// ============================================================
// Node MLP GEMM: C = act(A[M,512] @ Bpacked + bias)
//   SILU:   apply silu
//   RESID:  Cf32[m,n] += v   (residual stream, read-modify-write)
//   OUTF16: Ch[m,n] = (f16)v
// grid = (M/128, 512/128); 8 waves per block, wave = 16 rows x 128 cols.
// One A fragment per k-step feeds 8 WMMAs.
// ============================================================
template<int SILU, int RESID, int OUTF16>
__global__ __launch_bounds__(256, 1)
void node_gemm_k(const _Float16* __restrict__ A,
                 const _Float16* __restrict__ Bp,   // packed, KS=16 NT=32
                 const float* __restrict__ bias,
                 float* __restrict__ Cf,
                 _Float16* __restrict__ Ch) {
    const int lane = threadIdx.x & 31;
    const int wave = threadIdx.x >> 5;
    const int m0   = blockIdx.x * 128 + wave * 16;
    const int n0   = blockIdx.y * 128;
    const int ntb  = n0 >> 4;

    v8f acc[8] = {};
    const int r  = lane & 15;
    const int hi = lane >> 4;
    const _Float16* arow = A + (long)(m0 + r) * D_ + hi * 8;

    for (int ks = 0; ks < 16; ++ks) {
        AFrag ua;
        const _Float16* p = arow + ks * 32;
        ua.h[0] = *(const v8h*)(p);
        ua.h[1] = *(const v8h*)(p + 16);
#pragma unroll
        for (int j = 0; j < 8; ++j) {
            v16h b = load_b_frag(Bp, ks * 32 + ntb + j, lane);
            acc[j] = wmma_f16(ua.v, b, acc[j]);
        }
    }

#pragma unroll
    for (int j = 0; j < 8; ++j) {
        const int n     = n0 + j * 16 + r;
        const float bia = bias[n];
        const int mb    = m0 + 8 * hi;
#pragma unroll
        for (int q = 0; q < 8; ++q) {
            float v = acc[j][q] + bia;
            if (SILU) v = silu_f(v);
            const long idx = (long)(mb + q) * D_ + n;
            if (RESID) { v += Cf[idx]; Cf[idx] = v; }
            if (OUTF16) Ch[idx] = (_Float16)v;
        }
    }
}

// ============================================================
// host-side orchestration
// ============================================================
extern "C" void kernel_launch(void* const* d_in, const int* in_sizes, int n_in,
                              void* d_out, int out_size, void* d_ws, size_t ws_size,
                              hipStream_t stream) {
    const int*   tok    = (const int*)  d_in[0];
    const float* coords = (const float*)d_in[1];
    const int*   ei     = (const int*)  d_in[2];
    const float* emb    = (const float*)d_in[3];
    const float* ew1    = (const float*)d_in[4];
    const float* eb1    = (const float*)d_in[5];
    const float* ew2    = (const float*)d_in[6];
    const float* eb2    = (const float*)d_in[7];
    const float* nw1    = (const float*)d_in[8];
    const float* nb1    = (const float*)d_in[9];
    const float* nw2    = (const float*)d_in[10];
    const float* nb2    = (const float*)d_in[11];

    float* x    = (float*)d_out;                 // residual stream == encoder_rep
    float* mask = x + (long)BN_ * D_;            // padding mask (floats)

    char* ws = (char*)d_ws;
    _Float16* rbf  = (_Float16*)ws; ws += (size_t)E_  * RBF_ * sizeof(_Float16);
    _Float16* aggh = (_Float16*)ws; ws += (size_t)BN_ * D_   * sizeof(_Float16);
    _Float16* th   = (_Float16*)ws; ws += (size_t)BN_ * D_   * sizeof(_Float16);
    _Float16* ew1p = (_Float16*)ws; ws += (size_t)L_ * RBF_ * D_ * sizeof(_Float16);
    _Float16* ew2p = (_Float16*)ws; ws += (size_t)L_ * D_   * D_ * sizeof(_Float16);
    _Float16* nw1p = (_Float16*)ws; ws += (size_t)L_ * D_   * D_ * sizeof(_Float16);
    _Float16* nw2p = (_Float16*)ws; ws += (size_t)L_ * D_   * D_ * sizeof(_Float16);
    float*    dist = (float*)ws;    ws += (size_t)E_ * sizeof(float);

    // --- one-time (per call) weight packing into WMMA B-fragment order ---
    {
        const long t1 = (long)L_ * RBF_ * D_;
        const long t2 = (long)L_ * D_ * D_;
        pack_w_k<<<(int)((t1 + 255) / 256), 256, 0, stream>>>(ew1, ew1p, RBF_, D_, t1);
        pack_w_k<<<(int)((t2 + 255) / 256), 256, 0, stream>>>(ew2, ew2p, D_,   D_, t2);
        pack_w_k<<<(int)((t2 + 255) / 256), 256, 0, stream>>>(nw1, nw1p, D_,   D_, t2);
        pack_w_k<<<(int)((t2 + 255) / 256), 256, 0, stream>>>(nw2, nw2p, D_,   D_, t2);
    }

    // --- embedding, mask, distances, rbf features ---
    embed_k<<<(BN_ * D_) / 256, 256, 0, stream>>>(tok, emb, x);
    mask_k <<<BN_ / 256, 256, 0, stream>>>(tok, mask);
    dist_k <<<E_ / 256, 256, 0, stream>>>(ei, coords, dist);
    rbf_k  <<<(int)(((long)E_ * RBF_) / 256), 256, 0, stream>>>(dist, rbf);

    // --- interaction layers ---
    for (int l = 0; l < L_; ++l) {
        edge_layer_k<<<BN_, 256, 0, stream>>>(
            rbf, ei, x,
            ew1p + (long)l * RBF_ * D_, eb1 + (long)l * D_,
            ew2p + (long)l * D_ * D_,   eb2 + (long)l * D_,
            aggh);
        node_gemm_k<1, 0, 1><<<dim3(BN_ / 128, D_ / 128), 256, 0, stream>>>(
            aggh, nw1p + (long)l * D_ * D_, nb1 + (long)l * D_, nullptr, th);
        node_gemm_k<0, 1, 0><<<dim3(BN_ / 128, D_ / 128), 256, 0, stream>>>(
            th,   nw2p + (long)l * D_ * D_, nb2 + (long)l * D_, x, nullptr);
    }
}
